// RPGNN_90606630077048
// MI455X (gfx1250) — compile-verified, matrix operand order checked by reading, not run on hardware
//
#include <hip/hip_runtime.h>
#include <hip/hip_bf16.h>

// Problem constants (fixed by the reference)
#define NN   65536      // nodes
#define NE   262144     // edges
#define NG   2048       // graphs
#define DIN_ 9
#define HID  300
#define HID2 600
#define NL   5
#define FDIM 32
#define DGF  1523       // 300 + 32 + 1024 + 167

typedef __attribute__((ext_vector_type(16))) _Float16 v16h;
typedef __attribute__((ext_vector_type(8)))  float    v8f;

// ---------------------------------------------------------------------------
// f16-WMMA GEMM with LDS staging:  C[M,N] = act( A[M,K] @ B[K,N] + bias[N] )
// Block = 256 threads = 8 waves; macro-tile 128(M) x 64(N).
// Wave w owns rows [16w,16w+16) and all 64 columns: per 32-K slab it does one
// A register fill and FOUR WMMAs (A operand reused across 4 N-subtiles).
//   A slab: 128x32 f32 (16KB LDS) via global_load_async_to_lds_b128 fast path.
//   B slab: 32x64 f32 -> f16 transposed [col][k] (4KB LDS), coalesced float4.
// ---------------------------------------------------------------------------
__global__ __launch_bounds__(256) void k_gemm_wmma(
    const float* __restrict__ A, const float* __restrict__ B,
    const float* __restrict__ bias, float* __restrict__ C,
    int M, int K, int N, int do_relu)
{
    __shared__ __align__(16) float    ldsA[128 * 32];  // A slab, f32 [row][k]
    __shared__ __align__(16) _Float16 ldsB[64 * 32];   // B slab, f16 [col][k]

    const int tid  = threadIdx.x;
    const int wave = tid >> 5;
    const int lane = tid & 31;
    const int l15  = lane & 15;
    const int hi   = lane >> 4;

    const int  tilesN  = (N + 63) >> 6;
    const int  bm      = blockIdx.x / tilesN;
    const int  tn      = blockIdx.x % tilesN;
    const long rowBase = (long)bm * 128;
    const int  colBase = tn * 64;

    const bool vecA = ((K & 3) == 0) && (rowBase + 128 <= (long)M);
    const bool vecB = ((N & 3) == 0) && (colBase + 64 <= N);

    v8f acc[4];
#pragma unroll
    for (int s = 0; s < 4; ++s) acc[s] = (v8f){0.f,0.f,0.f,0.f,0.f,0.f,0.f,0.f};

    union HV { v16h v; _Float16 h[16]; float4 q[2]; };

    for (int kb = 0; kb < K; kb += 32) {
        const bool fullK = (kb + 32 <= K);

        // ------- stage A slab (128 rows x 32 k, f32) -------
        if (fullK && vecA) {
            // async DMA global -> LDS, 16B per op, 4 ops/thread (1024 total)
#pragma unroll
            for (int it = 0; it < 4; ++it) {
                const int chunk = tid + it * 256;       // 0..1023
                const int r  = chunk >> 3;              // row 0..127
                const int c4 = (chunk & 7) * 4;         // k offset 0..28
                unsigned ldsOff = (unsigned)(uintptr_t)(&ldsA[r * 32 + c4]);
                unsigned long long ga =
                    (unsigned long long)(uintptr_t)(&A[(rowBase + r) * (long)K + kb + c4]);
                asm volatile("global_load_async_to_lds_b128 %0, %1, off"
                             :: "v"(ldsOff), "v"(ga) : "memory");
            }
        } else {
            for (int chunk = tid; chunk < 128 * 32; chunk += 256) {
                const int r = chunk >> 5;
                const int k = chunk & 31;
                const long gr = rowBase + r;
                const int  gk = kb + k;
                ldsA[r * 32 + k] = (gr < (long)M && gk < K) ? A[gr * (long)K + gk] : 0.f;
            }
        }

        // ------- stage B slab (32 k x 64 cols -> f16 transposed [col][k]) ----
        if (fullK && vecB) {
            // 512 float4 chunks, 2 per thread; coalesced along columns
#pragma unroll
            for (int it = 0; it < 2; ++it) {
                const int chunk = tid + it * 256;       // 0..511
                const int k  = chunk >> 4;              // 0..31
                const int c4 = (chunk & 15) * 4;        // 0..60
                const float4 b4 = *(const float4*)&B[(long)(kb + k) * N + colBase + c4];
                ldsB[(c4 + 0) * 32 + k] = (_Float16)b4.x;
                ldsB[(c4 + 1) * 32 + k] = (_Float16)b4.y;
                ldsB[(c4 + 2) * 32 + k] = (_Float16)b4.z;
                ldsB[(c4 + 3) * 32 + k] = (_Float16)b4.w;
            }
        } else {
            for (int chunk = tid; chunk < 64 * 32; chunk += 256) {
                const int k = chunk & 31;
                const int c = chunk >> 5;               // 0..63
                const int gk = kb + k;
                const int gc = colBase + c;
                const float v = (gk < K && gc < N) ? B[(long)gk * N + gc] : 0.f;
                ldsB[c * 32 + k] = (_Float16)v;
            }
        }

        asm volatile("s_wait_asynccnt 0x0" ::: "memory");
        __syncthreads();

        // ------- register fill + 4x WMMA (A reused) -------
        // A layout (16-bit A 16x32): lane l15 = row; elements 0..7 <- K=8hi+0..7,
        // elements 8..15 <- K=16+8hi+0..7. B symmetric with lane l15 = column.
        HV ua;
        {
            const int rl = wave * 16 + l15;
            const float4* pa = (const float4*)&ldsA[rl * 32 + 8 * hi];
            const float4 a0 = pa[0], a1 = pa[1];   // K = 8hi .. 8hi+7
            const float4 a2 = pa[4], a3 = pa[5];   // K = 16+8hi .. +7
            ua.h[0]  = (_Float16)a0.x; ua.h[1]  = (_Float16)a0.y;
            ua.h[2]  = (_Float16)a0.z; ua.h[3]  = (_Float16)a0.w;
            ua.h[4]  = (_Float16)a1.x; ua.h[5]  = (_Float16)a1.y;
            ua.h[6]  = (_Float16)a1.z; ua.h[7]  = (_Float16)a1.w;
            ua.h[8]  = (_Float16)a2.x; ua.h[9]  = (_Float16)a2.y;
            ua.h[10] = (_Float16)a2.z; ua.h[11] = (_Float16)a2.w;
            ua.h[12] = (_Float16)a3.x; ua.h[13] = (_Float16)a3.y;
            ua.h[14] = (_Float16)a3.z; ua.h[15] = (_Float16)a3.w;
        }
#pragma unroll
        for (int s = 0; s < 4; ++s) {
            HV ub;
            const float4* pb = (const float4*)&ldsB[(16 * s + l15) * 32 + 8 * hi];
            ub.q[0] = pb[0];   // K = 8hi .. 8hi+7   (8 packed f16)
            ub.q[1] = pb[2];   // K = 16+8hi .. +7
            acc[s] = __builtin_amdgcn_wmma_f32_16x16x32_f16(
                         false, ua.v, false, ub.v, (short)0, acc[s], false, false);
        }
        __syncthreads();
    }

    // ------- epilogue: C/D layout M = vgpr + 8*hi, N = l15 -------
#pragma unroll
    for (int s = 0; s < 4; ++s) {
        const int c = colBase + 16 * s + l15;
        if (c < N) {
            const float bv = bias[c];
#pragma unroll
            for (int r = 0; r < 8; ++r) {
                const long m = rowBase + wave * 16 + r + 8 * hi;
                if (m < (long)M) {
                    float v = acc[s][r] + bv;
                    if (do_relu) v = fmaxf(v, 0.f);
                    C[m * (long)N + c] = v;
                }
            }
        }
    }
}

// ---------------------------------------------------------------------------
// Small helper kernels
// ---------------------------------------------------------------------------
__global__ void k_fill0(float* p, long n) {
    long i = (long)blockIdx.x * blockDim.x + threadIdx.x;
    if (i < n) p[i] = 0.f;
}
__global__ void k_fill0i(int* p, long n) {
    long i = (long)blockIdx.x * blockDim.x + threadIdx.x;
    if (i < n) p[i] = 0;
}
__global__ void k_copy(float* dst, const float* src, long n) {
    long i = (long)blockIdx.x * blockDim.x + threadIdx.x;
    if (i < n) dst[i] = src[i];
}
__global__ void k_addinto(float* dst, const float* src, long n) {
    long i = (long)blockIdx.x * blockDim.x + threadIdx.x;
    if (i < n) dst[i] += src[i];
}

// h = x @ W_emb + b_emb   (K=9: scalar loop)
__global__ void k_emb(const float* __restrict__ x, const float* __restrict__ W,
                      const float* __restrict__ b, float* __restrict__ h) {
    long i = (long)blockIdx.x * blockDim.x + threadIdx.x;
    if (i >= (long)NN * HID) return;
    int n = (int)(i / HID), j = (int)(i % HID);
    float s = b[j];
#pragma unroll
    for (int d = 0; d < DIN_; ++d) s += x[n * DIN_ + d] * W[d * HID + j];
    h[i] = s;
}

// vn[g][j] = vn0[j]
__global__ void k_bcast_vn(const float* vn0, float* vn) {
    long i = (long)blockIdx.x * blockDim.x + threadIdx.x;
    if (i >= (long)NG * HID) return;
    vn[i] = vn0[i % HID];
}

// hv = h + vn[batch]
__global__ void k_add_vn(const float* h, const float* vn, const int* batch, float* hv) {
    long i = (long)blockIdx.x * blockDim.x + threadIdx.x;
    if (i >= (long)NN * HID) return;
    int n = (int)(i / HID), j = (int)(i % HID);
    hv[i] = h[i] + vn[(long)batch[n] * HID + j];
}

// agg[dst] += hv[src]  (one block per edge)
__global__ void k_edge_scatter(const int* __restrict__ src, const int* __restrict__ dst,
                               const float* __restrict__ hv, float* __restrict__ agg) {
    int e = blockIdx.x;
    long s = (long)src[e] * HID;
    long d = (long)dst[e] * HID;
    for (int j = threadIdx.x; j < HID; j += blockDim.x)
        atomicAdd(&agg[d + j], hv[s + j]);
}

// z = (1+eps[l])*hv + agg   (may alias agg)
__global__ void k_combine(const float* hv, const float* agg, const float* eps, int l, float* z) {
    long i = (long)blockIdx.x * blockDim.x + threadIdx.x;
    if (i >= (long)NN * HID) return;
    z[i] = (1.f + eps[l]) * hv[i] + agg[i];
}

// per-column mean/var over M rows (block per column)
__global__ void k_colstats(const float* __restrict__ z, float* __restrict__ stats, int M) {
    __shared__ float ssum[256], ssq[256];
    int j = blockIdx.x;
    float s = 0.f, q = 0.f;
    for (int r = threadIdx.x; r < M; r += blockDim.x) {
        float v = z[(long)r * HID + j];
        s += v; q += v * v;
    }
    ssum[threadIdx.x] = s; ssq[threadIdx.x] = q;
    __syncthreads();
    for (int off = 128; off > 0; off >>= 1) {
        if ((int)threadIdx.x < off) {
            ssum[threadIdx.x] += ssum[threadIdx.x + off];
            ssq[threadIdx.x]  += ssq[threadIdx.x + off];
        }
        __syncthreads();
    }
    if (threadIdx.x == 0) {
        float m = ssum[0] / (float)M;
        stats[2 * j]     = m;
        stats[2 * j + 1] = ssq[0] / (float)M - m * m;
    }
}

// h += maybe_relu( bn(z2) )
__global__ void k_bn_apply(const float* __restrict__ z2, const float* __restrict__ stats,
                           const float* __restrict__ scale, const float* __restrict__ bias,
                           int l, int do_relu, float* __restrict__ h) {
    long i = (long)blockIdx.x * blockDim.x + threadIdx.x;
    if (i >= (long)NN * HID) return;
    int j = (int)(i % HID);
    float m = stats[2 * j], v = stats[2 * j + 1];
    float val = (z2[i] - m) * rsqrtf(v + 1e-5f) * scale[l * HID + j] + bias[l * HID + j];
    if (do_relu) val = fmaxf(val, 0.f);
    h[i] += val;
}

// seg-sum: out[batch[n]] += v[n]   (one block per node)
__global__ void k_node2graph(const int* __restrict__ batch, const float* __restrict__ v,
                             float* __restrict__ out) {
    int n = blockIdx.x;
    long g = (long)batch[n] * HID;
    long s = (long)n * HID;
    for (int j = threadIdx.x; j < HID; j += blockDim.x)
        atomicAdd(&out[g + j], v[s + j]);
}

__global__ void k_counts(const int* batch, int* cnt) {
    int n = blockIdx.x * blockDim.x + threadIdx.x;
    if (n < NN) atomicAdd(&cnt[batch[n]], 1);
}

__global__ void k_divcnt(float* pooled, const int* cnt) {
    long i = (long)blockIdx.x * blockDim.x + threadIdx.x;
    if (i >= (long)NG * HID) return;
    int g = (int)(i / HID);
    int c = cnt[g]; if (c < 1) c = 1;
    pooled[i] /= (float)c;
}

// h_rep = [pooled_h | id_pool(analytic) | morgan | maccs]
__global__ void k_hrep(const float* __restrict__ pooled, const int* __restrict__ cnt,
                       const float* __restrict__ morgan, const float* __restrict__ maccs,
                       float* __restrict__ hrep) {
    long i = (long)blockIdx.x * blockDim.x + threadIdx.x;
    if (i >= (long)NG * DGF) return;
    int g = (int)(i / DGF), j = (int)(i % DGF);
    float v;
    if (j < HID) {
        v = pooled[(long)g * HID + j];
    } else if (j < HID + FDIM) {
        // seg_mean(one_hot(rank % F)) is rank-set invariant: count of k in
        // [0,c) with k % F == f, divided by c. Independent of the RNG.
        int f = j - HID;
        int c = cnt[g];
        int cf = (c > f) ? ((c - 1 - f) / FDIM + 1) : 0;
        int cd = (c < 1) ? 1 : c;
        v = (float)cf / (float)cd;
    } else if (j < HID + FDIM + 1024) {
        v = morgan[(long)g * 1024 + (j - HID - FDIM)];
    } else {
        v = maccs[(long)g * 167 + (j - HID - FDIM - 1024)];
    }
    hrep[i] = v;
}

// out[g] = p1[g,:] . Wp2 + bp2
__global__ void k_final(const float* __restrict__ p1, const float* __restrict__ Wp2,
                        const float* __restrict__ bp2, float* __restrict__ out) {
    int g = blockIdx.x * blockDim.x + threadIdx.x;
    if (g >= NG) return;
    float s = bp2[0];
    const float* row = p1 + (long)g * HID2;
    for (int j = 0; j < HID2; ++j) s += row[j] * Wp2[j];
    out[g] = s;
}

// ---------------------------------------------------------------------------
// Orchestration
// ---------------------------------------------------------------------------
static inline int cdiv_l(long a, long b) { return (int)((a + b - 1) / b); }

static inline void launch_gemm(const float* A, const float* B, const float* bias,
                               float* C, int M, int K, int N, int relu, hipStream_t s) {
    int blocks = ((M + 127) / 128) * ((N + 63) / 64);
    k_gemm_wmma<<<blocks, 256, 0, s>>>(A, B, bias, C, M, K, N, relu);
}

extern "C" void kernel_launch(void* const* d_in, const int* in_sizes, int n_in,
                              void* d_out, int out_size, void* d_ws, size_t ws_size,
                              hipStream_t stream) {
    const float* x      = (const float*)d_in[0];
    const float* morgan = (const float*)d_in[1];
    const float* maccs  = (const float*)d_in[2];
    const int*   ei     = (const int*)d_in[3];    // [2,E]
    const int*   batch  = (const int*)d_in[4];
    const float* W_emb  = (const float*)d_in[5];
    const float* b_emb  = (const float*)d_in[6];
    const float* gW1    = (const float*)d_in[7];  // [L,300,600]
    const float* gb1    = (const float*)d_in[8];
    const float* gW2    = (const float*)d_in[9];  // [L,600,300]
    const float* gb2    = (const float*)d_in[10];
    const float* bns    = (const float*)d_in[11];
    const float* bnb    = (const float*)d_in[12];
    const float* eps    = (const float*)d_in[13];
    const float* vn0    = (const float*)d_in[14];
    const float* vW1    = (const float*)d_in[15]; // [4,300,600]
    const float* vb1    = (const float*)d_in[16];
    const float* vW2    = (const float*)d_in[17]; // [4,600,300]
    const float* vb2    = (const float*)d_in[18];
    const float* Wp1    = (const float*)d_in[19]; // [1523,600]
    const float* bp1    = (const float*)d_in[20];
    const float* Wp2    = (const float*)d_in[21]; // [600,1]
    const float* bp2    = (const float*)d_in[22];
    float* out = (float*)d_out;

    // workspace carve
    float* w = (float*)d_ws;
    float* h      = w;                       w += (long)NN * HID;
    float* hv     = w;                       w += (long)NN * HID;
    float* agg    = w;                       w += (long)NN * HID;   // also z
    float* t      = w;                       w += (long)NN * HID2;
    float* z2     = w;                       w += (long)NN * HID;
    float* vn     = w;                       w += (long)NG * HID;
    float* vt     = w;                       w += (long)NG * HID;
    float* vt1    = w;                       w += (long)NG * HID2;
    float* vdel   = w;                       w += (long)NG * HID;
    float* stats  = w;                       w += 2 * HID;
    float* pooled = w;                       w += (long)NG * HID;
    float* hrep   = w;                       w += (long)NG * DGF;
    float* p1     = w;                       w += (long)NG * HID2;
    int*   cnt    = (int*)w;

    const long NH  = (long)NN * HID;
    const long GH  = (long)NG * HID;
    const int  TPB = 256;

    // node embedding + vn init + counts
    k_emb<<<cdiv_l(NH, TPB), TPB, 0, stream>>>(x, W_emb, b_emb, h);
    k_bcast_vn<<<cdiv_l(GH, TPB), TPB, 0, stream>>>(vn0, vn);
    k_fill0i<<<cdiv_l(NG, TPB), TPB, 0, stream>>>(cnt, NG);
    k_counts<<<cdiv_l(NN, TPB), TPB, 0, stream>>>(batch, cnt);

    const int* esrc = ei;        // edge_index[0]
    const int* edst = ei + NE;   // edge_index[1]

    for (int l = 0; l < NL; ++l) {
        k_add_vn<<<cdiv_l(NH, TPB), TPB, 0, stream>>>(h, vn, batch, hv);
        k_fill0<<<cdiv_l(NH, TPB), TPB, 0, stream>>>(agg, NH);
        k_edge_scatter<<<NE, TPB, 0, stream>>>(esrc, edst, hv, agg);
        k_combine<<<cdiv_l(NH, TPB), TPB, 0, stream>>>(hv, agg, eps, l, agg);

        // GIN MLP: relu(z @ W1 + b1) @ W2 + b2   — WMMA GEMMs
        launch_gemm(agg, gW1 + (long)l * HID * HID2, gb1 + (long)l * HID2,
                    t, NN, HID, HID2, 1, stream);
        launch_gemm(t, gW2 + (long)l * HID2 * HID, gb2 + (long)l * HID,
                    z2, NN, HID2, HID, 0, stream);

        // batch-norm + residual
        k_colstats<<<HID, TPB, 0, stream>>>(z2, stats, NN);
        k_bn_apply<<<cdiv_l(NH, TPB), TPB, 0, stream>>>(z2, stats, bns, bnb,
                                                        l, (l < NL - 1) ? 1 : 0, h);

        // virtual-node update
        if (l < NL - 1) {
            k_copy<<<cdiv_l(GH, TPB), TPB, 0, stream>>>(vt, vn, GH);
            k_node2graph<<<NN, TPB, 0, stream>>>(batch, hv, vt);
            launch_gemm(vt, vW1 + (long)l * HID * HID2, vb1 + (long)l * HID2,
                        vt1, NG, HID, HID2, 1, stream);
            launch_gemm(vt1, vW2 + (long)l * HID2 * HID, vb2 + (long)l * HID,
                        vdel, NG, HID2, HID, 1, stream);
            k_addinto<<<cdiv_l(GH, TPB), TPB, 0, stream>>>(vn, vdel, GH);
        }
    }

    // pooled mean of h per graph
    k_fill0<<<cdiv_l(GH, TPB), TPB, 0, stream>>>(pooled, GH);
    k_node2graph<<<NN, TPB, 0, stream>>>(batch, h, pooled);
    k_divcnt<<<cdiv_l(GH, TPB), TPB, 0, stream>>>(pooled, cnt);

    // assemble h_rep = [pooled | id_pool | morgan | maccs]
    k_hrep<<<cdiv_l((long)NG * DGF, TPB), TPB, 0, stream>>>(pooled, cnt, morgan, maccs, hrep);

    // predictor
    launch_gemm(hrep, Wp1, bp1, p1, NG, DGF, HID2, 1, stream);
    k_final<<<cdiv_l(NG, TPB), TPB, 0, stream>>>(p1, Wp2, bp2, out);

    (void)in_sizes; (void)n_in; (void)out_size; (void)ws_size;
}
